// GraphMambaLayer_12378095747608
// MI455X (gfx1250) — matrix operand order repaired and checked by reading, not compiled, not run on hardware
//
#include <hip/hip_runtime.h>
#include <hip/hip_bf16.h>

typedef __attribute__((ext_vector_type(16))) _Float16 v16h;
typedef __attribute__((ext_vector_type(8)))  float    v8f;

#define GN   10000     // nodes
#define GD   128       // d_model
#define GK   16        // d_state
#define GS   5         // seq_len
#define GE   320000    // edges
#define GP   80000     // edges per tokenizer step (E/(S-1))

// Async copy of one 16-byte chunk global->LDS (gfx1250, tracked by ASYNCcnt).
__device__ __forceinline__ void async_copy_b128(unsigned lds_addr, unsigned long long gaddr) {
    asm volatile("global_load_async_to_lds_b128 %0, %1, off"
                 :: "v"(lds_addr), "v"(gaddr) : "memory");
}
__device__ __forceinline__ void async_wait0() {
    asm volatile("s_wait_asynccnt 0x0" ::: "memory");
}

// ---------------------------------------------------------------------------
// Utility: zero a float range
// ---------------------------------------------------------------------------
__global__ void zero_kernel(float* __restrict__ p, long n) {
    long i = (long)blockIdx.x * blockDim.x + threadIdx.x;
    if (i < n) p[i] = 0.0f;
}

// ---------------------------------------------------------------------------
// Convert x to f16 (GEMM A operand for the gate projection)
// ---------------------------------------------------------------------------
__global__ void convert_x_kernel(const float* __restrict__ x,
                                 _Float16* __restrict__ x_h, long n) {
    long i = (long)blockIdx.x * blockDim.x + threadIdx.x;
    if (i < n) x_h[i] = (_Float16)x[i];
}

// ---------------------------------------------------------------------------
// Pack weights to f16, column-major (col*128 + k) so each lane's K-run is
// contiguous for the WMMA B fragment.
// WcatT: 160 cols = [W_delta(128) | W_B(16) | W_C(16)]
// ---------------------------------------------------------------------------
__global__ void prep_weights(const float* __restrict__ W_delta,
                             const float* __restrict__ W_B,
                             const float* __restrict__ W_C,
                             const float* __restrict__ W_g,
                             const float* __restrict__ W_out,
                             _Float16* __restrict__ WcatT,
                             _Float16* __restrict__ WgT,
                             _Float16* __restrict__ WoutT) {
    int k = blockIdx.x;    // 0..127  input-dim
    int c = threadIdx.x;   // 0..159  output col
    float v;
    if (c < 128)      v = W_delta[k * 128 + c];
    else if (c < 144) v = W_B[k * 16 + (c - 128)];
    else              v = W_C[k * 16 + (c - 144)];
    WcatT[(size_t)c * 128 + k] = (_Float16)v;
    if (c < 128) {
        WgT  [(size_t)c * 128 + k] = (_Float16)W_g  [k * 128 + c];
        WoutT[(size_t)c * 128 + k] = (_Float16)W_out[k * 128 + c];
    }
}

// ---------------------------------------------------------------------------
// Edge scatter: one block (128 lanes) per sampled edge; lane d handles dim d.
// msg = x[src] + rel_table[type]; atomic add into seg[step][dst][:] and deg.
// x (5MB) + rel_table (100KB) are L2-resident, so random gathers are L2 hits.
// ---------------------------------------------------------------------------
__global__ void scatter_kernel(const float* __restrict__ x,
                               const int* __restrict__ edge_index,
                               const int* __restrict__ edge_type,
                               const int* __restrict__ perms,
                               const float* __restrict__ rel_table,
                               float* __restrict__ seg,
                               float* __restrict__ deg) {
    int b    = blockIdx.x;            // 0 .. E-1
    int step = b / GP;                // 0..3
    int p    = b - step * GP;
    int e    = perms[step * GP + p];
    int src  = edge_index[e];
    int dst  = edge_index[GE + e];
    int typ  = edge_type[e];
    int d    = threadIdx.x;
    float msg = x[(size_t)src * GD + d] + rel_table[(size_t)typ * GD + d];
    atomicAdd(&seg[(((size_t)step * GN + dst) * GD) + d], msg);
    if (d == 0) atomicAdd(&deg[(size_t)step * GN + dst], 1.0f);
}

// ---------------------------------------------------------------------------
// Token assembly: tokens[n][0]=x[n]; tokens[n][s]=seg[s-1][n]/max(deg,1).
// Also emits the f16 copy used as the WMMA A operand.
// ---------------------------------------------------------------------------
__global__ void token_kernel(const float* __restrict__ x,
                             const float* __restrict__ seg,
                             const float* __restrict__ deg,
                             float* __restrict__ tokens,
                             _Float16* __restrict__ tokens_h) {
    int n = blockIdx.x;
    int d = threadIdx.x;
    float v0 = x[(size_t)n * GD + d];
    tokens  [((size_t)n * GS + 0) * GD + d] = v0;
    tokens_h[((size_t)n * GS + 0) * GD + d] = (_Float16)v0;
#pragma unroll
    for (int s = 1; s < GS; ++s) {
        float dg = fmaxf(deg[(size_t)(s - 1) * GN + n], 1.0f);
        float v  = seg[(((size_t)(s - 1) * GN + n) * GD) + d] / dg;
        tokens  [((size_t)n * GS + s) * GD + d] = v;
        tokens_h[((size_t)n * GS + s) * GD + d] = (_Float16)v;
    }
}

// ---------------------------------------------------------------------------
// GEMM #1: [50000,128] @ [128,160] via v_wmma_f32_16x16x32_f16.
// Block = 10 waves; wave w owns output cols [16w,16w+16).
// A-tile (16x128 f16 = one contiguous 4KB block) staged to LDS with
// global_load_async_to_lds_b128 (ASYNCcnt), drained with s_wait_asynccnt.
// Epilogue: softplus(+b_delta) for delta cols, raw store for B/C cols.
// ---------------------------------------------------------------------------
__global__ __launch_bounds__(320) void gemm_tokens_kernel(
        const _Float16* __restrict__ tokens_h,
        const _Float16* __restrict__ WcatT,
        const float* __restrict__ b_delta,
        float* __restrict__ deltas,
        float* __restrict__ Bs,
        float* __restrict__ Cs) {
    __shared__ __align__(16) _Float16 As[16 * 128];
    int tid = threadIdx.x;
    int m0  = blockIdx.x * 16;
    if (tid < 256) {   // 256 lanes x 16B = whole 4KB tile
        unsigned long long g =
            (unsigned long long)(uintptr_t)(tokens_h + (size_t)m0 * GD) +
            (unsigned long long)tid * 16ull;
        unsigned l = (unsigned)(uintptr_t)(&As[0]) + (unsigned)tid * 16u;
        async_copy_b128(l, g);
    }
    async_wait0();
    __syncthreads();

    int lane = tid & 31, w = tid >> 5;
    int cl = lane & 15, hi = lane >> 4;
    int c  = w * 16 + cl;
    const _Float16* Wc = WcatT + (size_t)c * 128;

    v8f acc = {};
#pragma unroll
    for (int kk = 0; kk < 128; kk += 32) {
        v16h a, b;
#pragma unroll
        for (int j = 0; j < 8; ++j) {
            int ka = 2 * j + hi * 16 + kk;                             // A: contiguous k = kk + 16*hi + s
            a[2 * j]     = As[cl * 128 + ka];
            a[2 * j + 1] = As[cl * 128 + ka + 1];
            int kb = 2 * j + hi * 16 + kk;                             // B: same contiguous k order as A
            b[2 * j]     = Wc[kb];
            b[2 * j + 1] = Wc[kb + 1];
        }
        acc = __builtin_amdgcn_wmma_f32_16x16x32_f16(
                  false, a, false, b, (short)0, acc, false, false);
    }

    int mb = hi * 8;
#pragma unroll
    for (int r = 0; r < 8; ++r) {
        int row = m0 + mb + r;         // flattened (n*S+s)
        float v = acc[r];
        if (c < 128) {
            float t = v + b_delta[c];
            deltas[(size_t)row * GD + c] = (t > 20.0f) ? t : log1pf(__expf(t));
        } else if (c < 144) {
            Bs[(size_t)row * GK + (c - 128)] = v;
        } else {
            Cs[(size_t)row * GK + (c - 144)] = v;
        }
    }
}

// ---------------------------------------------------------------------------
// Bidirectional selective scan. Block = 128 lanes = one node; lane owns dim d,
// keeps state[16] in VGPRs. exp-heavy (16 v_exp per (s,dir) per lane).
// log_A (8KB) staged coalesced into LDS. Output written f16 (only feeds the
// W_out WMMA). mean_s(out_f + rev(out_b)) == (sum_f + sum_b)/S.
// ---------------------------------------------------------------------------
__global__ __launch_bounds__(128) void scan_kernel(
        const float* __restrict__ tokens,
        const float* __restrict__ deltas,
        const float* __restrict__ Bs,
        const float* __restrict__ Cs,
        const float* __restrict__ log_A,
        _Float16* __restrict__ scan_out_h) {
    int n = blockIdx.x;
    int d = threadIdx.x;
    __shared__ float sB[GS * GK], sC[GS * GK];
    __shared__ float sA[GD * GK];
    if (d < GS * GK) {
        sB[d] = Bs[(size_t)n * GS * GK + d];
        sC[d] = Cs[(size_t)n * GS * GK + d];
    }
    for (int i = d; i < GD * GK; i += GD) sA[i] = log_A[i];
    __syncthreads();

    float Ac[GK];
#pragma unroll
    for (int k = 0; k < GK; ++k) Ac[k] = -__expf(sA[d * GK + k]);

    float tok[GS], del[GS];
#pragma unroll
    for (int s = 0; s < GS; ++s) {
        tok[s] = tokens[((size_t)n * GS + s) * GD + d];
        del[s] = deltas[((size_t)n * GS + s) * GD + d];
    }

    float acc = 0.0f;
    float st[GK];
#pragma unroll
    for (int k = 0; k < GK; ++k) st[k] = 0.0f;
#pragma unroll
    for (int s = 0; s < GS; ++s) {            // forward
        float dl = del[s], tk = tok[s];
#pragma unroll
        for (int k = 0; k < GK; ++k) {
            float a = __expf(dl * Ac[k]);
            st[k] = a * st[k] + dl * sB[s * GK + k] * tk;
            acc += st[k] * sC[s * GK + k];
        }
    }
#pragma unroll
    for (int k = 0; k < GK; ++k) st[k] = 0.0f;
#pragma unroll
    for (int s = GS - 1; s >= 0; --s) {       // backward
        float dl = del[s], tk = tok[s];
#pragma unroll
        for (int k = 0; k < GK; ++k) {
            float a = __expf(dl * Ac[k]);
            st[k] = a * st[k] + dl * sB[s * GK + k] * tk;
            acc += st[k] * sC[s * GK + k];
        }
    }
    scan_out_h[(size_t)n * GD + d] = (_Float16)(acc * (1.0f / GS));
}

// ---------------------------------------------------------------------------
// GEMM #2 (dual) + SiLU gate + residual + LayerNorm.
// Block = 8 waves = 16 rows x full 128 cols; wave w owns cols [16w,16w+16).
// Both A-tiles (x_h, scan_out_h; contiguous 4KB each) staged via async-to-LDS.
// g = x@W_g, o = scan_out@W_out with two WMMA accumulators, then
// res = x + silu(g+bg)*(o+bo) staged to LDS for the row-wide LN reduction.
// ---------------------------------------------------------------------------
__global__ __launch_bounds__(256) void final_kernel(
        const float* __restrict__ x,
        const _Float16* __restrict__ x_h,
        const _Float16* __restrict__ scan_out_h,
        const _Float16* __restrict__ WgT,
        const _Float16* __restrict__ WoutT,
        const float* __restrict__ b_g,
        const float* __restrict__ b_out,
        const float* __restrict__ ln_g,
        const float* __restrict__ ln_b,
        float* __restrict__ out) {
    __shared__ __align__(16) _Float16 Ax[2048], Asc[2048];
    __shared__ float res[16 * 128];
    __shared__ float mu[16], rs[16];

    int tid = threadIdx.x;
    int m0  = blockIdx.x * 16;
    {
        unsigned long long gx =
            (unsigned long long)(uintptr_t)(x_h + (size_t)m0 * GD) +
            (unsigned long long)tid * 16ull;
        unsigned long long gs =
            (unsigned long long)(uintptr_t)(scan_out_h + (size_t)m0 * GD) +
            (unsigned long long)tid * 16ull;
        unsigned lx = (unsigned)(uintptr_t)(&Ax[0])  + (unsigned)tid * 16u;
        unsigned ls = (unsigned)(uintptr_t)(&Asc[0]) + (unsigned)tid * 16u;
        async_copy_b128(lx, gx);
        async_copy_b128(ls, gs);
    }
    async_wait0();
    __syncthreads();

    int lane = tid & 31, w = tid >> 5;
    int cl = lane & 15, hi = lane >> 4;
    int c  = w * 16 + cl;
    const _Float16* Wg = WgT   + (size_t)c * 128;
    const _Float16* Wo = WoutT + (size_t)c * 128;

    v8f accg = {}, acco = {};
#pragma unroll
    for (int kk = 0; kk < 128; kk += 32) {
        v16h ax, as2, bg, bo;
#pragma unroll
        for (int j = 0; j < 8; ++j) {
            int ka = 2 * j + hi * 16 + kk;
            ax [2 * j]     = Ax [cl * 128 + ka];
            ax [2 * j + 1] = Ax [cl * 128 + ka + 1];
            as2[2 * j]     = Asc[cl * 128 + ka];
            as2[2 * j + 1] = Asc[cl * 128 + ka + 1];
            int kb = 2 * j + hi * 16 + kk;
            bg[2 * j]     = Wg[kb];
            bg[2 * j + 1] = Wg[kb + 1];
            bo[2 * j]     = Wo[kb];
            bo[2 * j + 1] = Wo[kb + 1];
        }
        accg = __builtin_amdgcn_wmma_f32_16x16x32_f16(
                   false, ax,  false, bg, (short)0, accg, false, false);
        acco = __builtin_amdgcn_wmma_f32_16x16x32_f16(
                   false, as2, false, bo, (short)0, acco, false, false);
    }

    float bgc = b_g[c], boc = b_out[c];
    int mb = hi * 8;
#pragma unroll
    for (int r = 0; r < 8; ++r) {
        int m = mb + r;
        float g = accg[r] + bgc;
        float o = acco[r] + boc;
        float sil = g / (1.0f + __expf(-g));
        res[m * 128 + c] = x[(size_t)(m0 + m) * GD + c] + sil * o;
    }
    __syncthreads();

    if (tid < 16) {
        float s1 = 0.0f, s2 = 0.0f;
        for (int j = 0; j < 128; ++j) {
            float v = res[tid * 128 + j];
            s1 += v; s2 += v * v;
        }
        float m_ = s1 * (1.0f / 128.0f);
        mu[tid] = m_;
        rs[tid] = rsqrtf(s2 * (1.0f / 128.0f) - m_ * m_ + 1e-5f);
    }
    __syncthreads();

    for (int i = tid; i < 2048; i += 256) {
        int m = i >> 7, cc = i & 127;
        out[(size_t)(m0 + m) * GD + cc] =
            (res[i] - mu[m]) * rs[m] * ln_g[cc] + ln_b[cc];
    }
}

// ---------------------------------------------------------------------------
// Launch
// ---------------------------------------------------------------------------
extern "C" void kernel_launch(void* const* d_in, const int* in_sizes, int n_in,
                              void* d_out, int out_size, void* d_ws, size_t ws_size,
                              hipStream_t stream) {
    const float* x          = (const float*)d_in[0];
    const int*   edge_index = (const int*)  d_in[1];
    const int*   edge_type  = (const int*)  d_in[2];
    const int*   perms      = (const int*)  d_in[3];
    const float* rel_table  = (const float*)d_in[4];
    const float* log_A      = (const float*)d_in[5];
    const float* W_B        = (const float*)d_in[6];
    const float* W_C        = (const float*)d_in[7];
    const float* W_delta    = (const float*)d_in[8];
    const float* b_delta    = (const float*)d_in[9];
    const float* W_g        = (const float*)d_in[10];
    const float* b_g        = (const float*)d_in[11];
    const float* W_out      = (const float*)d_in[12];
    const float* b_out      = (const float*)d_in[13];
    const float* ln_g       = (const float*)d_in[14];
    const float* ln_b       = (const float*)d_in[15];
    float* out = (float*)d_out;

    // ---- workspace carve-up (256B aligned) ----
    char* ws = (char*)d_ws;
    size_t off = 0;
    #define TAKE(ptr_t, name, bytes)                                   \
        ptr_t name = (ptr_t)(ws + off);                                \
        off = (off + (size_t)(bytes) + 255) & ~(size_t)255;

    TAKE(float*,      seg,        (size_t)(GS - 1) * GN * GD * 4)   // 20.5 MB
    TAKE(float*,      deg,        (size_t)(GS - 1) * GN * 4)
    TAKE(float*,      tokens,     (size_t)GN * GS * GD * 4)         // 25.6 MB
    TAKE(_Float16*,   tokens_h,   (size_t)GN * GS * GD * 2)         // 12.8 MB
    TAKE(float*,      deltas,     (size_t)GN * GS * GD * 4)         // 25.6 MB
    TAKE(float*,      Bs,         (size_t)GN * GS * GK * 4)
    TAKE(float*,      Cs,         (size_t)GN * GS * GK * 4)
    TAKE(_Float16*,   x_h,        (size_t)GN * GD * 2)
    TAKE(_Float16*,   scan_out_h, (size_t)GN * GD * 2)
    TAKE(_Float16*,   WcatT,      (size_t)160 * 128 * 2)
    TAKE(_Float16*,   WgT,        (size_t)128 * 128 * 2)
    TAKE(_Float16*,   WoutT,      (size_t)128 * 128 * 2)
    #undef TAKE
    (void)ws_size; (void)in_sizes; (void)n_in; (void)out_size;

    // 1) zero the scatter accumulators
    {
        long nseg = (long)(GS - 1) * GN * GD;
        zero_kernel<<<(unsigned)((nseg + 255) / 256), 256, 0, stream>>>(seg, nseg);
        long ndeg = (long)(GS - 1) * GN;
        zero_kernel<<<(unsigned)((ndeg + 255) / 256), 256, 0, stream>>>(deg, ndeg);
    }
    // 2) pack weights to f16 column-major; convert x to f16
    prep_weights<<<128, 160, 0, stream>>>(W_delta, W_B, W_C, W_g, W_out,
                                          WcatT, WgT, WoutT);
    {
        long nx = (long)GN * GD;
        convert_x_kernel<<<(unsigned)((nx + 255) / 256), 256, 0, stream>>>(x, x_h, nx);
    }
    // 3) edge scatter
    scatter_kernel<<<GE, GD, 0, stream>>>(x, edge_index, edge_type, perms,
                                          rel_table, seg, deg);
    // 4) token assembly (f32 + f16 copies)
    token_kernel<<<GN, GD, 0, stream>>>(x, seg, deg, tokens, tokens_h);
    // 5) fused delta/B/C projection (WMMA, async-to-LDS A staging) + softplus
    gemm_tokens_kernel<<<(GN * GS) / 16, 320, 0, stream>>>(
        tokens_h, WcatT, b_delta, deltas, Bs, Cs);
    // 6) bidirectional selective scan
    scan_kernel<<<GN, GD, 0, stream>>>(tokens, deltas, Bs, Cs, log_A, scan_out_h);
    // 7) gate/out projections (WMMA, async-to-LDS) + SiLU + residual + LayerNorm
    final_kernel<<<GN / 16, 256, 0, stream>>>(
        x, x_h, scan_out_h, WgT, WoutT, b_g, b_out, ln_g, ln_b, out);
}
